// PyGGCNModel_26096221290527
// MI455X (gfx1250) — matrix operand order repaired and checked by reading, not compiled, not run on hardware
//
#include <hip/hip_runtime.h>
#include <stdint.h>

#define FDIM 128
#define LN_EPS 1e-5f

typedef __attribute__((ext_vector_type(16))) _Float16 v16h;
typedef __attribute__((ext_vector_type(8)))  _Float16 v8h;
typedef __attribute__((ext_vector_type(8)))  float    v8f;

__device__ __forceinline__ float fix_x(float v) {
    // nan_to_num(nan=0, posinf=0, neginf=0)
    if (isnan(v) || isinf(v)) return 0.0f;
    return v;
}
__device__ __forceinline__ float fix_ew(float v) {
    // nan_to_num(nan=0, posinf=1, neginf=0)
    if (isnan(v)) return 0.0f;
    if (isinf(v)) return v > 0.0f ? 1.0f : 0.0f;
    return v;
}

// ---- LayerNorm: one wave (32 lanes) per row, float4 per lane, shfl reduce ----
__global__ void ln_kernel(const float* __restrict__ x, const float* __restrict__ g,
                          const float* __restrict__ b, _Float16* __restrict__ xh, int n) {
    int row = blockIdx.x * 8 + (threadIdx.x >> 5);
    if (row >= n) return;
    int lane = threadIdx.x & 31;
    float4 v4 = ((const float4*)(x + (size_t)row * FDIM))[lane];
    float a0 = fix_x(v4.x), a1 = fix_x(v4.y), a2 = fix_x(v4.z), a3 = fix_x(v4.w);
    float s = a0 + a1 + a2 + a3;
    float q = a0 * a0 + a1 * a1 + a2 * a2 + a3 * a3;
    #pragma unroll
    for (int off = 16; off > 0; off >>= 1) {
        s += __shfl_xor(s, off, 32);
        q += __shfl_xor(q, off, 32);
    }
    float mu  = s * (1.0f / FDIM);
    float var = q * (1.0f / FDIM) - mu * mu;
    float rs  = rsqrtf(var + LN_EPS);
    int c = lane * 4;
    _Float16* o = xh + (size_t)row * FDIM + c;
    o[0] = (_Float16)((a0 - mu) * rs * g[c + 0] + b[c + 0]);
    o[1] = (_Float16)((a1 - mu) * rs * g[c + 1] + b[c + 1]);
    o[2] = (_Float16)((a2 - mu) * rs * g[c + 2] + b[c + 2]);
    o[3] = (_Float16)((a3 - mu) * rs * g[c + 3] + b[c + 3]);
}

// ---- transpose + convert W1,W2 (128x128 f32) -> Wt (f16, column-major rows) ----
__global__ void prep_w_kernel(const float* __restrict__ W1, const float* __restrict__ W2,
                              _Float16* __restrict__ Wt) {
    int t = blockIdx.x * blockDim.x + threadIdx.x;   // 0..32767
    if (t >= 2 * FDIM * FDIM) return;
    int which = t >> 14;
    int i = t & 16383;
    int k = i >> 7, n = i & 127;
    const float* W = which ? W2 : W1;
    _Float16* T = Wt + (size_t)which * FDIM * FDIM;
    T[(size_t)n * FDIM + k] = (_Float16)W[(size_t)k * FDIM + n];
}

// ---- degree with self loop: deg[i] = 1 + sum ew over incoming edges ----
__global__ void deg_init_kernel(float* __restrict__ deg, int n) {
    int i = blockIdx.x * blockDim.x + threadIdx.x;
    if (i < n) deg[i] = 1.0f;
}
__global__ void deg_scatter_kernel(const long long* __restrict__ col,
                                   const float* __restrict__ ea,
                                   float* __restrict__ deg, int e) {
    int i = blockIdx.x * blockDim.x + threadIdx.x;
    if (i >= e) return;
    atomicAdd(&deg[(int)col[i]], fix_ew(ea[i]));
}
__global__ void dis_kernel(float* __restrict__ deg_dis, float* __restrict__ selfnorm, int n) {
    int i = blockIdx.x * blockDim.x + threadIdx.x;
    if (i >= n) return;
    float d = deg_dis[i];
    float dis = d > 0.0f ? rsqrtf(d) : 0.0f;
    deg_dis[i] = dis;              // in-place: deg -> dis
    selfnorm[i] = dis * dis;       // self-loop norm (ew = 1)
}
__global__ void norm_kernel(const long long* __restrict__ row, const long long* __restrict__ col,
                            const float* __restrict__ ea, const float* __restrict__ dis,
                            float* __restrict__ nw, int e) {
    int i = blockIdx.x * blockDim.x + threadIdx.x;
    if (i >= e) return;
    nw[i] = dis[(int)row[i]] * fix_ew(ea[i]) * dis[(int)col[i]];
}

// ---- GEMM: C[M x 128] = A_f16[M x 128] @ W (via Wt f16 [128n x 128k]) ----
// one wave per 16x16 tile; block of 256 = 8 waves = one 16-row block x all 128 cols
__global__ void gemm_f16_kernel(const _Float16* __restrict__ A, const _Float16* __restrict__ Bt,
                                float* __restrict__ C, int M) {
    int wave = threadIdx.x >> 5;
    int lane = threadIdx.x & 31;
    int m0 = blockIdx.x * 16;
    if (m0 + 16 > M) return;
    int n0 = wave * 16;
    int lo16 = lane & 15;
    int hi = lane >> 4;

    const _Float16* arow = A + (size_t)(m0 + lo16) * FDIM;   // A row for this lane (M = lane%16)
    const _Float16* brow = Bt + (size_t)(n0 + lo16) * FDIM;  // W column (as Wt row) for this lane

    v8f acc = {};
    #pragma unroll
    for (int kk = 0; kk < FDIM; kk += 32) {
        // A 16x32 f16 layout: lane<16 -> K {0..7,16..23}; lane>=16 -> K {8..15,24..31}
        int ka = kk + hi * 8;
        v8h alo = *(const v8h*)(arow + ka);
        v8h ahi = *(const v8h*)(arow + ka + 16);
        v16h a = __builtin_shufflevector(alo, ahi, 0, 1, 2, 3, 4, 5, 6, 7,
                                                   8, 9, 10, 11, 12, 13, 14, 15);
        // B 32x16 f16 layout: lane<16 -> K {0..7,16..23} of column N=lane%16; lane>=16 -> K {8..15,24..31}
        int kb = kk + hi * 8;
        v8h blo = *(const v8h*)(brow + kb);
        v8h bhi = *(const v8h*)(brow + kb + 16);
        v16h bb = __builtin_shufflevector(blo, bhi, 0, 1, 2, 3, 4, 5, 6, 7,
                                                    8, 9, 10, 11, 12, 13, 14, 15);
        acc = __builtin_amdgcn_wmma_f32_16x16x32_f16(false, a, false, bb,
                                                     (short)0, acc, false, false);
    }
    // D layout: VGPR r holds C[m0 + r + 8*hi][n0 + lane%16]
    float* crow = C + (size_t)(m0 + hi * 8) * FDIM + n0 + lo16;
    #pragma unroll
    for (int r = 0; r < 8; ++r) crow[(size_t)r * FDIM] = acc[r];
}

// ---- agg init: agg = selfnorm * h + bias  (self loop + bias) ----
__global__ void init_agg_kernel(const float* __restrict__ h, const float* __restrict__ selfnorm,
                                const float* __restrict__ bias, float* __restrict__ agg, int n) {
    size_t idx = (size_t)blockIdx.x * blockDim.x + threadIdx.x;
    if (idx >= (size_t)n * FDIM) return;
    int f = (int)(idx & (FDIM - 1));
    int node = (int)(idx >> 7);
    agg[idx] = selfnorm[node] * h[idx] + bias[f];
}

// ---- edge gather-scale-scatter: one wave per edge, float4 per lane ----
__global__ void edge_agg_kernel(const float* __restrict__ h, const long long* __restrict__ row,
                                const long long* __restrict__ col, const float* __restrict__ nw,
                                float* __restrict__ agg, int e) {
    int eid = blockIdx.x * 8 + (threadIdx.x >> 5);
    if (eid >= e) return;
    float w = nw[eid];
    if (w == 0.0f) return;  // uniform across wave
    int lane = threadIdx.x & 31;
    int r = (int)row[eid], c = (int)col[eid];
    float4 v = ((const float4*)(h + (size_t)r * FDIM))[lane];
    float* dst = agg + (size_t)c * FDIM + lane * 4;
    atomicAdd(dst + 0, v.x * w);
    atomicAdd(dst + 1, v.y * w);
    atomicAdd(dst + 2, v.z * w);
    atomicAdd(dst + 3, v.w * w);
}

// ---- relu in place (f32) + f16 copy for next GEMM ----
__global__ void relu_cvt_kernel(float* __restrict__ agg, _Float16* __restrict__ xh, int n) {
    size_t idx = (size_t)blockIdx.x * blockDim.x + threadIdx.x;
    if (idx >= (size_t)n * FDIM) return;
    float v = agg[idx];
    v = v > 0.0f ? v : 0.0f;
    agg[idx] = v;
    xh[idx] = (_Float16)v;
}

// ---- output layer: h3 = x2 @ W3 ; out = selfnorm*h3 + b3 (edges scattered after) ----
__global__ void out_layer_kernel(const float* __restrict__ x2, const float* __restrict__ W3,
                                 const float* __restrict__ b3, const float* __restrict__ selfnorm,
                                 float* __restrict__ h3, float* __restrict__ out, int n) {
    int node = blockIdx.x * 8 + (threadIdx.x >> 5);
    if (node >= n) return;
    int lane = threadIdx.x & 31;
    float4 v = ((const float4*)(x2 + (size_t)node * FDIM))[lane];
    float4 w = ((const float4*)W3)[lane];
    float s = v.x * w.x + v.y * w.y + v.z * w.z + v.w * w.w;
    #pragma unroll
    for (int off = 16; off > 0; off >>= 1) s += __shfl_xor(s, off, 32);
    if (lane == 0) {
        h3[node] = s;
        out[node] = selfnorm[node] * s + b3[0];
    }
}
__global__ void edge_out_kernel(const long long* __restrict__ row, const long long* __restrict__ col,
                                const float* __restrict__ nw, const float* __restrict__ h3,
                                float* __restrict__ out, int e) {
    int i = blockIdx.x * blockDim.x + threadIdx.x;
    if (i >= e) return;
    float w = nw[i];
    if (w == 0.0f) return;
    atomicAdd(&out[(int)col[i]], h3[(int)row[i]] * w);
}

extern "C" void kernel_launch(void* const* d_in, const int* in_sizes, int n_in,
                              void* d_out, int out_size, void* d_ws, size_t ws_size,
                              hipStream_t stream) {
    const float*     x     = (const float*)d_in[0];
    const long long* ei    = (const long long*)d_in[1];   // int64 (2,E)
    const float*     ea    = (const float*)d_in[2];       // (E,1)
    const float*     gamma = (const float*)d_in[3];
    const float*     beta  = (const float*)d_in[4];
    const float*     W1    = (const float*)d_in[5];
    const float*     b1    = (const float*)d_in[6];
    const float*     W2    = (const float*)d_in[7];
    const float*     b2    = (const float*)d_in[8];
    const float*     W3    = (const float*)d_in[9];
    const float*     b3    = (const float*)d_in[10];

    const int N = in_sizes[0] / FDIM;
    const int E = in_sizes[2];
    const long long* rowp = ei;
    const long long* colp = ei + E;

    // workspace carving (256B aligned)
    char* ws = (char*)d_ws;
    size_t off = 0;
    auto carve = [&](size_t bytes) -> char* {
        char* p = ws + off;
        off = (off + bytes + 255) & ~(size_t)255;
        return p;
    };
    float*     deg_dis  = (float*)carve((size_t)N * 4);            // deg then dis
    float*     selfnorm = (float*)carve((size_t)N * 4);
    float*     nw       = (float*)carve((size_t)E * 4);
    _Float16*  xh       = (_Float16*)carve((size_t)N * FDIM * 2);
    float*     h        = (float*)carve((size_t)N * FDIM * 4);
    float*     agg      = (float*)carve((size_t)N * FDIM * 4);
    _Float16*  Wt       = (_Float16*)carve((size_t)2 * FDIM * FDIM * 2);
    float*     h3       = (float*)carve((size_t)N * 4);
    (void)ws_size; (void)n_in; (void)out_size;

    float* out = (float*)d_out;

    const int B = 256;
    int gN    = (N + B - 1) / B;
    int gE    = (E + B - 1) / B;
    int gNW   = (N + 7) / 8;                 // wave-per-row kernels
    int gEW   = (E + 7) / 8;                 // wave-per-edge kernels
    int gNF   = (int)(((size_t)N * FDIM + B - 1) / B);
    int gGemm = N / 16;

    // preprocessing
    prep_w_kernel<<<(2 * FDIM * FDIM + B - 1) / B, B, 0, stream>>>(W1, W2, Wt);
    ln_kernel<<<gNW, B, 0, stream>>>(x, gamma, beta, xh, N);
    deg_init_kernel<<<gN, B, 0, stream>>>(deg_dis, N);
    deg_scatter_kernel<<<gE, B, 0, stream>>>(colp, ea, deg_dis, E);
    dis_kernel<<<gN, B, 0, stream>>>(deg_dis, selfnorm, N);
    norm_kernel<<<gE, B, 0, stream>>>(rowp, colp, ea, deg_dis, nw, E);

    // layer 1
    gemm_f16_kernel<<<gGemm, B, 0, stream>>>(xh, Wt, h, N);
    init_agg_kernel<<<gNF, B, 0, stream>>>(h, selfnorm, b1, agg, N);
    edge_agg_kernel<<<gEW, B, 0, stream>>>(h, rowp, colp, nw, agg, E);
    relu_cvt_kernel<<<gNF, B, 0, stream>>>(agg, xh, N);

    // layer 2
    gemm_f16_kernel<<<gGemm, B, 0, stream>>>(xh, Wt + FDIM * FDIM, h, N);
    init_agg_kernel<<<gNF, B, 0, stream>>>(h, selfnorm, b2, agg, N);
    edge_agg_kernel<<<gEW, B, 0, stream>>>(h, rowp, colp, nw, agg, E);
    relu_cvt_kernel<<<gNF, B, 0, stream>>>(agg, xh, N);

    // output layer (H -> 1)
    out_layer_kernel<<<gNW, B, 0, stream>>>(agg, W3, b3, selfnorm, h3, out, N);
    edge_out_kernel<<<gE, B, 0, stream>>>(rowp, colp, nw, h3, out, E);
}
